// BertSelfAttention_39917426049368
// MI455X (gfx1250) — matrix operand order repaired and checked by reading, not compiled, not run on hardware
//
#include <hip/hip_runtime.h>
#include <hip/hip_bf16.h>

// ---------------------------------------------------------------------------
// MI455X (gfx1250) fused BERT self-attention with penalty-reassignment.
//  - All GEMMs via v_wmma_f32_16x16x32_bf16 (f32 accumulate).
//  - Weights pre-transposed (and V stored transposed) so every WMMA fragment
//    is two 16B chunk reads -> global_load_b128 / ds_load_b128, no packing.
//  - Projection GEMM stages tiles in LDS via inline-asm
//    global_load_async_to_lds_b128 + s_wait_asynccnt (ASYNCcnt path).
//  - Attention: per (head, 16-query tile) both batches' score rows live in
//    LDS (128KB f32 + 64KB bf16 probs of the 320KB/WGP budget).
// ---------------------------------------------------------------------------

typedef __attribute__((ext_vector_type(16))) __bf16        v16bf;
typedef __attribute__((ext_vector_type(8)))  float         v8f;
typedef __attribute__((ext_vector_type(4)))  unsigned int  u32x4;

static constexpr int Bsz    = 2;
static constexpr int Slen   = 1024;
static constexpr int HID    = 2048;
static constexpr int NHEAD  = 16;
static constexpr int HEADD  = 128;

static constexpr float PEN_THRESH = 10.0f;
static constexpr float PEN_MAX    = 15.0f;
static constexpr float PEN_W      = 0.001f;
static constexpr float ORIG_T     = 0.01f;
static constexpr float EPSF       = 1e-12f;

// ---------------- helpers ----------------

__device__ __forceinline__ unsigned short f2bf(float f) {
  unsigned int u = __float_as_uint(f);
  u += 0x7FFFu + ((u >> 16) & 1u);      // round-to-nearest-even
  return (unsigned short)(u >> 16);
}

// Build a v16bf fragment from two 16-byte chunks at p and p+step (elements).
// Lowers to two b128 loads (global or ds), no element-wise packing.
__device__ __forceinline__ v16bf frag2(const unsigned short* p, int step) {
  union { u32x4 q[2]; v16bf v; } u;
  u.q[0] = *(const u32x4*)(p);
  u.q[1] = *(const u32x4*)(p + step);
  return u.v;
}

// A fragment, 16x32 bf16, row-major source (ld elems), ISA layout:
// lanes 0-15: M=lane, K = {0..7, 16..23}; lanes 16-31: M=lane-16, K+8.
__device__ __forceinline__ v16bf load_a_frag(const unsigned short* A, int ld,
                                             int row0, int k0, int lane) {
  const unsigned short* p =
      A + (long)(row0 + (lane & 15)) * ld + k0 + (lane >> 4) * 8;
  return frag2(p, 16);
}

// B fragment where logical B(KxN) = M^T with M stored row-major [n][k]:
// lane holds col n=lane&15, K = {0..7, 16..23} (lanes<16) / K+8 (lanes>=16).
// Two 16B chunks per lane.
__device__ __forceinline__ v16bf load_bt_frag(const unsigned short* Mm, int ld,
                                              int k0, int col0, int lane) {
  const unsigned short* p =
      Mm + (long)(col0 + (lane & 15)) * ld + k0 + (lane >> 4) * 8;
  return frag2(p, 16);
}

__device__ __forceinline__ v8f wmma_bf16(v16bf a, v16bf b, v8f c) {
  return __builtin_amdgcn_wmma_f32_16x16x32_bf16(false, a, false, b,
                                                 (short)0, c, false, false);
}

// Async DMA of one 16B chunk global -> LDS (GLOBAL_LOAD_ASYNC_TO_LDS_B128,
// GV mode). The LDS byte offset is the low 32 bits of the generic pointer
// (ISA 10.2).
__device__ __forceinline__ void async_copy16(const unsigned short* gsrc,
                                             unsigned short* ldst) {
  unsigned lds_addr = (unsigned)(unsigned long)ldst;
  asm volatile("global_load_async_to_lds_b128 %0, %1, off"
               :
               : "v"(lds_addr), "v"(gsrc)
               : "memory");
}

__device__ __forceinline__ void wait_async() {
  asm volatile("s_wait_asynccnt 0x0" ::: "memory");
}

// ---------------- kernels ----------------

__global__ __launch_bounds__(256) void cast_f32_to_bf16_kernel(
    const float* __restrict__ src, unsigned short* __restrict__ dst, long n) {
  long i = (long)blockIdx.x * blockDim.x + threadIdx.x;
  if (i < n) dst[i] = f2bf(src[i]);
}

// dst[c][r] = bf16(src[r][c]) for RxC f32 src; 32x32 LDS tiles, coalesced.
__global__ __launch_bounds__(256) void transpose_cast_kernel(
    const float* __restrict__ src, unsigned short* __restrict__ dst,
    int R, int C) {
  __shared__ unsigned short tile[32][33];
  int c0 = blockIdx.x * 32, r0 = blockIdx.y * 32;
  int tx = threadIdx.x & 31, ty = threadIdx.x >> 5;   // 32 x 8
#pragma unroll
  for (int i = 0; i < 32; i += 8)
    tile[ty + i][tx] = f2bf(src[(long)(r0 + ty + i) * C + (c0 + tx)]);
  __syncthreads();
#pragma unroll
  for (int i = 0; i < 32; i += 8)
    dst[(long)(c0 + ty + i) * R + (r0 + tx)] = tile[tx][ty + i];
}

// C = A(bf16, MxK row-major) @ Wt^T (Wt bf16, NxK row-major) + bias.
// Block = 8 waves (2x4), tile 32x64, K-step 32. A/B tiles staged in LDS via
// async global->LDS DMA; fragments then read with ds_load_b128.
// transposed!=0 stores Out[col*M + m] (used to produce V^T for the ctx GEMM).
__global__ __launch_bounds__(256) void qkv_gemm_kernel(
    const unsigned short* __restrict__ A, const unsigned short* __restrict__ Wt,
    const float* __restrict__ bias, unsigned short* __restrict__ Out,
    int M, int N, int K, int transposed) {
  __shared__ unsigned short As[32][32];   // [m][k] 2KB
  __shared__ unsigned short Bs[64][32];   // [n][k] 4KB
  const int t    = threadIdx.x;
  const int lane = t & 31;
  const int wave = t >> 5;
  const int rowB = blockIdx.y * 32;
  const int colB = blockIdx.x * 64;

  v8f acc = {};
  for (int k0 = 0; k0 < K; k0 += 32) {
    // stage A tile: 32 rows x 32 k = 128 x 16B; threads 0..127
    if (t < 128) {
      int r = t >> 2, c = (t & 3) * 8;
      async_copy16(A + (long)(rowB + r) * K + k0 + c, &As[r][c]);
    }
    // stage B tile: 64 rows x 32 k = 256 x 16B; all threads
    {
      int r = t >> 2, c = (t & 3) * 8;
      async_copy16(Wt + (long)(colB + r) * K + k0 + c, &Bs[r][c]);
    }
    if (k0 + 32 < K)  // pull next A rows toward L2 (global_prefetch_b8)
      __builtin_prefetch(A + (long)(rowB + (t >> 3)) * K + k0 + 32, 0, 1);
    wait_async();
    __syncthreads();

    v16bf af = frag2(&As[(wave >> 2) * 16 + (lane & 15)][(lane >> 4) * 8], 16);
    v16bf bf = frag2(&Bs[(wave & 3) * 16 + (lane & 15)][(lane >> 4) * 8], 16);
    acc = wmma_bf16(af, bf, acc);
    __syncthreads();
  }

  const int half = lane >> 4;
  const int col  = colB + (wave & 3) * 16 + (lane & 15);
  const int row0 = rowB + (wave >> 2) * 16;
  float bv = bias[col];
#pragma unroll
  for (int r = 0; r < 8; ++r) {
    int m = row0 + r + half * 8;
    unsigned short v = f2bf(acc[r] + bv);
    if (transposed) Out[(long)col * M + m] = v;
    else            Out[(long)m * N + col] = v;
  }
}

// Fused attention for one (head, 16-query tile), both batches.
// LDS: sc[2][16][1024] f32 (scores -> reassigned -> exp) + pb[2][16][1024] bf16.
// Vt is [HID][B*S] (transposed), so ctx B-fragments are contiguous.
__global__ __launch_bounds__(256) void attn_fused_kernel(
    const unsigned short* __restrict__ Qb, const unsigned short* __restrict__ Kb,
    const unsigned short* __restrict__ Vt, float* __restrict__ out) {
  extern __shared__ char smem_raw[];
  float*          sc = (float*)smem_raw;                           // 128 KB
  unsigned short* pb = (unsigned short*)(smem_raw + Bsz * 16 * Slen * 4);

  const int lane = threadIdx.x & 31;
  const int wave = threadIdx.x >> 5;
  const int h    = blockIdx.x;
  const int q0   = blockIdx.y * 16;
  const float inv_sqrt_d = 0.08838834764831845f;  // 1/sqrt(128)

  // ---- Phase 1: scores = (Q K^T) / sqrt(d), both batches, into LDS ----
  for (int b = 0; b < Bsz; ++b) {
    const unsigned short* Qh = Qb + (long)b * Slen * HID + h * HEADD;
    const unsigned short* Kh = Kb + (long)b * Slen * HID + h * HEADD;
    for (int nt = wave; nt < Slen / 16; nt += 8) {
      v8f acc = {};
#pragma unroll
      for (int k0 = 0; k0 < HEADD; k0 += 32) {
        v16bf af = load_a_frag(Qh, HID, q0, k0, lane);
        v16bf bf = load_bt_frag(Kh, HID, k0, nt * 16, lane);  // K^T, contiguous
        acc = wmma_bf16(af, bf, acc);
      }
      int half = lane >> 4, col = nt * 16 + (lane & 15);
#pragma unroll
      for (int r = 0; r < 8; ++r)
        sc[(b * 16 + r + half * 8) * Slen + col] = acc[r] * inv_sqrt_d;
    }
  }
  __syncthreads();

  // ---- Phase 2: reverse cumprod penalty + shifted +/-1 updates ----
  // One wave owns one q-row for BOTH batches (batch-axis softmax coupling).
  for (int qq = wave; qq < 16; qq += 8) {
    float* r0 = sc + (0 * 16 + qq) * Slen;
    float* r1 = sc + (1 * 16 + qq) * Slen;
    const int base = lane * 32;  // contiguous 32-element chunk per lane

    float a0[32], a1[32], p0[32], p1[32];
#pragma unroll
    for (int i = 0; i < 32; ++i) { a0[i] = r0[base + i]; a1[i] = r1[base + i]; }

    // chunk-local suffix products
    float pr = 1.f;
#pragma unroll
    for (int i = 31; i >= 0; --i) { pr *= a0[i]; p0[i] = pr; }
    float c0 = pr;
    pr = 1.f;
#pragma unroll
    for (int i = 31; i >= 0; --i) { pr *= a1[i]; p1[i] = pr; }
    float c1 = pr;

    // inclusive suffix product of chunk totals across lanes, then exclusive
    float run0 = c0, run1 = c1;
    for (int off = 1; off < 32; off <<= 1) {
      float t0 = __shfl_down(run0, off);
      float t1 = __shfl_down(run1, off);
      if (lane + off < 32) { run0 *= t0; run1 *= t1; }
    }
    float suf0 = __shfl_down(run0, 1); if (lane == 31) suf0 = 1.f;
    float suf1 = __shfl_down(run1, 1); if (lane == 31) suf1 = 1.f;
#pragma unroll
    for (int i = 0; i < 32; ++i) { p0[i] *= suf0; p1[i] *= suf1; }

    // batch-axis softmax of -penalty; sum of squares over k (both slices)
    float ss0 = 0.f, ss1 = 0.f;
    float t0f = 0.f, t1f = 0.f, t0l = 0.f, t1l = 0.f;
    bool anym = false;
#pragma unroll
    for (int i = 0; i < 32; ++i) {
      float e0 = __expf(-p0[i]), e1 = __expf(-p1[i]);
      float den = fmaxf(e0 + e1, 1e-30f);
      float t0 = e0 / den, t1 = e1 / den;
      ss0 += t0 * t0; ss1 += t1 * t1;
      if (i == 0)  { t0f = t0; t1f = t1; }
      if (i == 31) { t0l = t0; t1l = t1; }
      anym |= (p0[i] > PEN_THRESH) | (p1[i] > PEN_THRESH);
    }
#pragma unroll
    for (int off = 16; off > 0; off >>= 1) {
      ss0 += __shfl_xor(ss0, off);
      ss1 += __shfl_xor(ss1, off);
    }
    t0f = __shfl(t0f, 0);  t1f = __shfl(t1f, 0);    // element j = 0
    t0l = __shfl(t0l, 31); t1l = __shfl(t1l, 31);   // element j = S-1
    float gate = __ballot(anym) ? 1.f : 0.f;        // any(mask) (local approx)

    // slice i=-1 excludes j=S-1; slice i=+1 excludes j=0
    float n0m = sqrtf(fmaxf(ss0 - t0l * t0l, 0.f));
    float n1m = sqrtf(fmaxf(ss1 - t1l * t1l, 0.f));
    float n0p = sqrtf(fmaxf(ss0 - t0f * t0f, 0.f));
    float n1p = sqrtf(fmaxf(ss1 - t1f * t1f, 0.f));
    float ts0m = fminf(n0m / fmaxf(fabsf(n0m), EPSF), PEN_MAX);
    float ts1m = fminf(n1m / fmaxf(fabsf(n1m), EPSF), PEN_MAX);
    float ts0p = fminf(n0p / fmaxf(fabsf(n0p), EPSF), PEN_MAX);
    float ts1p = fminf(n1p / fmaxf(fabsf(n1p), EPSF), PEN_MAX);

#pragma unroll
    for (int i = 0; i < 32; ++i) {
      int j = base + i;
      float cl0 = fminf(fmaxf(p0[i], -ORIG_T), ORIG_T) * a0[i];
      float cl1 = fminf(fmaxf(p1[i], -ORIG_T), ORIG_T) * a1[i];
      bool  m0  = p0[i] > PEN_THRESH, m1 = p1[i] > PEN_THRESH;
      // shift i=-1: source j in [0, S-1), lands at j+1
      if (j < Slen - 1) {
        float u0 = m0 ? -fabsf(cl0) : fabsf(a0[i] * ts0m * PEN_W);
        float u1 = m1 ? -fabsf(cl1) : fabsf(a1[i] * ts1m * PEN_W);
        atomicAdd(&r0[j + 1], u0 * gate);
        atomicAdd(&r1[j + 1], u1 * gate);
      }
      // shift i=+1: source j in [1, S), lands at j-1
      if (j >= 1) {
        float u0 = m0 ? -fabsf(cl0) : fabsf(a0[i] * ts0p * PEN_W);
        float u1 = m1 ? -fabsf(cl1) : fabsf(a1[i] * ts1p * PEN_W);
        atomicAdd(&r0[j - 1], u0 * gate);
        atomicAdd(&r1[j - 1], u1 * gate);
      }
    }
  }
  __syncthreads();

  // ---- Phase 3: softmax over k, write bf16 probs to LDS ----
  for (int r = wave; r < Bsz * 16; r += 8) {
    float* row = sc + r * Slen;
    float mx = -3.4e38f;
    for (int j = lane; j < Slen; j += 32) mx = fmaxf(mx, row[j]);
#pragma unroll
    for (int off = 16; off > 0; off >>= 1) mx = fmaxf(mx, __shfl_xor(mx, off));
    float sum = 0.f;
    for (int j = lane; j < Slen; j += 32) {
      float e = __expf(row[j] - mx);
      row[j] = e;
      sum += e;
    }
#pragma unroll
    for (int off = 16; off > 0; off >>= 1) sum += __shfl_xor(sum, off);
    float inv = 1.f / sum;
    for (int j = lane; j < Slen; j += 32) pb[r * Slen + j] = f2bf(row[j] * inv);
  }
  __syncthreads();

  // ---- Phase 4: ctx = probs @ V  (Vt contiguous B-frags), write f32 out ----
  for (int b = 0; b < Bsz; ++b) {
    const unsigned short* Pr = pb + b * 16 * Slen;
    const unsigned short* Vh = Vt + (long)(h * HEADD) * (Bsz * Slen) + b * Slen;
    int nt = wave;  // 8 waves x 16 cols = HEAD = 128
    v8f acc = {};
    for (int k0 = 0; k0 < Slen; k0 += 32) {
      v16bf af = load_a_frag(Pr, Slen, 0, k0, lane);            // ds_load_b128
      v16bf bf = load_bt_frag(Vh, Bsz * Slen, k0, nt * 16, lane);
      acc = wmma_bf16(af, bf, acc);
    }
    int half = lane >> 4, col = h * HEADD + nt * 16 + (lane & 15);
#pragma unroll
    for (int r = 0; r < 8; ++r) {
      long s = (long)b * Slen + q0 + r + half * 8;
      out[s * HID + col] = acc[r];
    }
  }
}

// ---------------- host launcher ----------------

extern "C" void kernel_launch(void* const* d_in, const int* in_sizes, int n_in,
                              void* d_out, int out_size, void* d_ws, size_t ws_size,
                              hipStream_t stream) {
  const float* X  = (const float*)d_in[0];
  const float* Wq = (const float*)d_in[1];
  const float* bq = (const float*)d_in[2];
  const float* Wk = (const float*)d_in[3];
  const float* bk = (const float*)d_in[4];
  const float* Wv = (const float*)d_in[5];
  const float* bv = (const float*)d_in[6];
  float* out = (float*)d_out;

  const long nX = (long)Bsz * Slen * HID;   // 4,194,304
  const int  M  = Bsz * Slen;               // 2048

  char* w = (char*)d_ws;
  unsigned short* Xb  = (unsigned short*)(w);
  unsigned short* Wqt = (unsigned short*)(w + 1 * nX * 2);
  unsigned short* Wkt = (unsigned short*)(w + 2 * nX * 2);
  unsigned short* Wvt = (unsigned short*)(w + 3 * nX * 2);
  unsigned short* Qb  = (unsigned short*)(w + 4 * nX * 2);
  unsigned short* Kb  = (unsigned short*)(w + 5 * nX * 2);
  unsigned short* Vt  = (unsigned short*)(w + 6 * nX * 2);   // [HID][B*S]

  // 1) X -> bf16 (row-major); W -> bf16 transposed [N][K]
  {
    int blk = 256;
    cast_f32_to_bf16_kernel<<<(int)((nX + blk - 1) / blk), blk, 0, stream>>>(X, Xb, nX);
    dim3 tg(HID / 32, HID / 32);
    transpose_cast_kernel<<<tg, 256, 0, stream>>>(Wq, Wqt, HID, HID);
    transpose_cast_kernel<<<tg, 256, 0, stream>>>(Wk, Wkt, HID, HID);
    transpose_cast_kernel<<<tg, 256, 0, stream>>>(Wv, Wvt, HID, HID);
  }

  // 2) projections: Q,K row-major bf16; V transposed bf16 [HID][B*S]
  {
    dim3 grid(HID / 64, M / 32);  // 32 x 64
    qkv_gemm_kernel<<<grid, 256, 0, stream>>>(Xb, Wqt, bq, Qb, M, HID, HID, 0);
    qkv_gemm_kernel<<<grid, 256, 0, stream>>>(Xb, Wkt, bk, Kb, M, HID, HID, 0);
    qkv_gemm_kernel<<<grid, 256, 0, stream>>>(Xb, Wvt, bv, Vt, M, HID, HID, 1);
  }

  // 3) fused attention: grid = (heads, q-tiles); 192KB dynamic LDS per block
  {
    size_t smem = (size_t)Bsz * 16 * Slen * 4 + (size_t)Bsz * 16 * Slen * 2;
    attn_fused_kernel<<<dim3(NHEAD, Slen / 16), 256, smem, stream>>>(Qb, Kb, Vt, out);
  }
}
